// Net_SLSTM_TempAtten_74268574482511
// MI455X (gfx1250) — matrix operand order repaired and checked by reading, not compiled, not run on hardware
//
#include <hip/hip_runtime.h>
#include <hip/hip_bf16.h>
#include <cmath>

typedef __attribute__((ext_vector_type(16))) _Float16 v16h;
typedef __attribute__((ext_vector_type(8)))  _Float16 v8h;
typedef __attribute__((ext_vector_type(8)))  float    v8f;

#define Bsz 256
#define Hsz 512
#define Isz 14
#define Ipad 32
#define Tsz 512
#define G4H 2048

// ---------------------------------------------------------------------------
// WMMA helpers (gfx1250: D = A(16x32 f16) x B(32x16 f16) + C(16x16 f32))
// ---------------------------------------------------------------------------
__device__ __forceinline__ v8f wmma_f16(v16h a, v16h b, v8f c) {
  return __builtin_amdgcn_wmma_f32_16x16x32_f16(
      /*neg_a=*/false, a, /*neg_b=*/false, b,
      /*c_mod=*/(short)0, c, /*reuse_a=*/false, /*reuse_b=*/false);
}

// A-matrix 16x32 f16: lane = 16*(khalf)+M ; halves 0..15 hold
// K = kbase..kbase+15, kbase = (lane>>4)*16 (matches B's contiguous K order).
__device__ __forceinline__ v16h load_a16(const _Float16* __restrict__ p,
                                         int stride, int row, int k0) {
  const _Float16* r = p + row * stride + k0;
  v8h lo = *(const v8h*)(r);
  v8h hi = *(const v8h*)(r + 8);
  v16h a;
#pragma unroll
  for (int i = 0; i < 8; ++i) { a[i] = lo[i]; a[i + 8] = hi[i]; }
  return a;
}

// B-matrix 32x16 f16 (KxN): lane = 16*(khalf)+N ; halves 0..15 hold
// K = (lane>>4)*16 + 0..15.  B[k][n] = W[n][k] => contiguous K run of W row n.
__device__ __forceinline__ v16h load_b16(const _Float16* __restrict__ p,
                                         int stride, int row, int k0) {
  return *(const v16h*)(p + row * stride + k0);
}

__device__ __forceinline__ float sigf(float x) {
  return 1.0f / (1.0f + __expf(-x));
}
// robust fast tanh: saturates to +/-1 for large |x| (expf(inf)=inf -> 1-0)
__device__ __forceinline__ float tanh_fast(float x) {
  return 1.0f - 2.0f / (__expf(2.0f * x) + 1.0f);
}

// ---------------------------------------------------------------------------
// Layer 1: gates = xpad_t @ Wih1p^T + mem1 @ W_hh1^T + b ; fused SLSTM
// ---------------------------------------------------------------------------
__global__ __launch_bounds__(256) void slstm_layer1(
    const _Float16* __restrict__ xpad,     // [B, 32] f16 (zero-padded x_t)
    const _Float16* __restrict__ Wihp,     // [4H, 32] f16 (zero-padded)
    const _Float16* __restrict__ Whh,      // [4H, H] f16
    const float*    __restrict__ bias,     // [4H] (b_ih + b_hh)
    const float*    __restrict__ thr_p,    // scalar
    const _Float16* __restrict__ memh_in,  // [B, H] f16 (A operand, prev step)
    float*          __restrict__ syn,      // [B, H] f32 in/out
    float*          __restrict__ mem,      // [B, H] f32 in/out
    _Float16*       __restrict__ memh_out, // [B, H] f16 (A for next step)
    _Float16*       __restrict__ spkh)     // [B, H] f16 spikes (0/1)
{
  const int lane  = threadIdx.x & 31;
  const int wave  = threadIdx.x >> 5;
  const int mtile = blockIdx.y * 8 + wave;   // 16 rows of B
  const int ntile = blockIdx.x;              // 16 cols of H
  const int lhi   = lane >> 4;
  const int llo   = lane & 15;

  v8f acc[4];
#pragma unroll
  for (int g = 0; g < 4; ++g) { v8f z = {}; acc[g] = z; }

  // ---- input term: one 16x16x32 WMMA per gate (pre-padded, branch-free) ----
  {
    v16h a = load_a16(xpad, Ipad, mtile * 16 + llo, lhi * 16);
#pragma unroll
    for (int g = 0; g < 4; ++g) {
      v16h b = load_b16(Wihp, Ipad, g * Hsz + ntile * 16 + llo, lhi * 16);
      acc[g] = wmma_f16(a, b, acc[g]);
    }
  }

  // ---- recurrent term: mem1 @ W_hh^T, K = 512 ----
#pragma unroll 2
  for (int kk = 0; kk < Hsz; kk += 32) {
    v16h a = load_a16(memh_in, Hsz, mtile * 16 + llo, kk + lhi * 16);
#pragma unroll
    for (int g = 0; g < 4; ++g) {
      v16h b = load_b16(Whh, Hsz, g * Hsz + ntile * 16 + llo, kk + lhi * 16);
      acc[g] = wmma_f16(a, b, acc[g]);
    }
  }

  // ---- fused SLSTM elementwise on the 16x16 tile ----
  const float thr  = *thr_p;
  const int   ncol = ntile * 16 + llo;
  const float bi = bias[0 * Hsz + ncol];
  const float bf = bias[1 * Hsz + ncol];
  const float bg = bias[2 * Hsz + ncol];
  const float bo = bias[3 * Hsz + ncol];
#pragma unroll
  for (int j = 0; j < 8; ++j) {
    const int m   = mtile * 16 + lhi * 8 + j;   // C/D layout: M = j + 8*(lane>=16)
    const int idx = m * Hsz + ncol;
    float ig = sigf(acc[0][j] + bi);
    float fg = sigf(acc[1][j] + bf);
    float gg = tanh_fast(acc[2][j] + bg);
    float og = sigf(acc[3][j] + bo);
    float c  = fg * syn[idx] + ig * gg;
    float mo = mem[idx];
    float hh = og * tanh_fast(c);
    float mn = hh - ((mo > thr) ? thr : 0.0f);  // detached subtract-reset
    syn[idx]      = c;
    mem[idx]      = mn;
    memh_out[idx] = (_Float16)mn;
    spkh[idx]     = (mn > thr) ? (_Float16)1.0f : (_Float16)0.0f;
  }
}

// ---------------------------------------------------------------------------
// Layer 2: gates = spk1 @ W_ih2^T + mem2 @ W_hh2^T + b ; fused elementwise +
// attenuation-weighted accumulation of mem2 into accum.
// ---------------------------------------------------------------------------
__global__ __launch_bounds__(256) void slstm_layer2(
    const _Float16* __restrict__ spk_in,   // [B, H] f16
    const _Float16* __restrict__ Wih,      // [4H, H] f16
    const _Float16* __restrict__ Whh,      // [4H, H] f16
    const float*    __restrict__ bias,     // [4H]
    const float*    __restrict__ thr_p,
    const _Float16* __restrict__ memh_in,  // [B, H] f16
    float*          __restrict__ syn,      // [B, H] f32 in/out
    float*          __restrict__ mem,      // [B, H] f32 in/out
    _Float16*       __restrict__ memh_out, // [B, H] f16
    float*          __restrict__ accum,    // [B, H] f32 weighted sum
    float wt)
{
  const int lane  = threadIdx.x & 31;
  const int wave  = threadIdx.x >> 5;
  const int mtile = blockIdx.y * 8 + wave;
  const int ntile = blockIdx.x;
  const int lhi   = lane >> 4;
  const int llo   = lane & 15;

  v8f acc[4];
#pragma unroll
  for (int g = 0; g < 4; ++g) { v8f z = {}; acc[g] = z; }

#pragma unroll 2
  for (int kk = 0; kk < Hsz; kk += 32) {
    v16h ax = load_a16(spk_in,  Hsz, mtile * 16 + llo, kk + lhi * 16);
    v16h ah = load_a16(memh_in, Hsz, mtile * 16 + llo, kk + lhi * 16);
#pragma unroll
    for (int g = 0; g < 4; ++g) {
      const int row = g * Hsz + ntile * 16 + llo;
      v16h bx = load_b16(Wih, Hsz, row, kk + lhi * 16);
      acc[g] = wmma_f16(ax, bx, acc[g]);
      v16h bh = load_b16(Whh, Hsz, row, kk + lhi * 16);
      acc[g] = wmma_f16(ah, bh, acc[g]);
    }
  }

  const float thr  = *thr_p;
  const int   ncol = ntile * 16 + llo;
  const float bi = bias[0 * Hsz + ncol];
  const float bf = bias[1 * Hsz + ncol];
  const float bg = bias[2 * Hsz + ncol];
  const float bo = bias[3 * Hsz + ncol];
#pragma unroll
  for (int j = 0; j < 8; ++j) {
    const int m   = mtile * 16 + lhi * 8 + j;
    const int idx = m * Hsz + ncol;
    float ig = sigf(acc[0][j] + bi);
    float fg = sigf(acc[1][j] + bf);
    float gg = tanh_fast(acc[2][j] + bg);
    float og = sigf(acc[3][j] + bo);
    float c  = fg * syn[idx] + ig * gg;
    float mo = mem[idx];
    float hh = og * tanh_fast(c);
    float mn = hh - ((mo > thr) ? thr : 0.0f);
    syn[idx]      = c;
    mem[idx]      = mn;
    memh_out[idx] = (_Float16)mn;
    accum[idx]   += wt * mn;   // each (m,n) owned by exactly one lane: race-free
  }
}

// ---------------------------------------------------------------------------
// One-shot per launch: f32 -> f16 weight conversion + bias fusion + zero-pad
// of x (K:14->32 for all T) and W_ih1 (K:14->32).
// ---------------------------------------------------------------------------
__global__ void prep_weights(
    const float* __restrict__ x,                                     // [T*B, 14]
    const float* __restrict__ Wih1, const float* __restrict__ Whh1,
    const float* __restrict__ bih1, const float* __restrict__ bhh1,
    const float* __restrict__ Wih2, const float* __restrict__ Whh2,
    const float* __restrict__ bih2, const float* __restrict__ bhh2,
    _Float16* __restrict__ Xpad,                                     // [T*B, 32]
    _Float16* __restrict__ Wih1p,                                    // [4H, 32]
    _Float16* __restrict__ Whh1h,
    _Float16* __restrict__ Wih2h, _Float16* __restrict__ Whh2h,
    float* __restrict__ b1, float* __restrict__ b2)
{
  const int NX = Tsz * Bsz * Ipad;   // 4,194,304 (largest task)
  const int NW = G4H * Hsz;          // 1,048,576
  for (int i = blockIdx.x * blockDim.x + threadIdx.x; i < NX;
       i += gridDim.x * blockDim.x) {
    {
      int row = i >> 5, k = i & 31;
      Xpad[i] = (k < Isz) ? (_Float16)x[row * Isz + k] : (_Float16)0.0f;
    }
    if (i < NW) {
      Whh1h[i] = (_Float16)Whh1[i];
      Wih2h[i] = (_Float16)Wih2[i];
      Whh2h[i] = (_Float16)Whh2[i];
    }
    if (i < G4H * Ipad) {
      int row = i >> 5, k = i & 31;
      Wih1p[i] = (k < Isz) ? (_Float16)Wih1[row * Isz + k] : (_Float16)0.0f;
    }
    if (i < G4H) { b1[i] = bih1[i] + bhh1[i]; b2[i] = bih2[i] + bhh2[i]; }
  }
}

__global__ void zero_f32(float* __restrict__ p, int n) {
  int i = blockIdx.x * blockDim.x + threadIdx.x;
  if (i < n) p[i] = 0.0f;
}

// ---------------------------------------------------------------------------
// Final FC: out[b,c] = sum_h accum[b,h] * W_fc[c,h] + b_fc[c]   (256x8, tiny)
// ---------------------------------------------------------------------------
__global__ void fc_out(const float* __restrict__ accum,
                       const float* __restrict__ Wfc,
                       const float* __restrict__ bfc,
                       float* __restrict__ out)
{
  int tid = blockIdx.x * blockDim.x + threadIdx.x;
  if (tid >= Bsz * 8) return;
  int b = tid >> 3, c = tid & 7;
  const float* ar = accum + b * Hsz;
  const float* wr = Wfc + c * Hsz;
  float s = bfc[c];
  for (int h = 0; h < Hsz; ++h) s += ar[h] * wr[h];
  out[tid] = s;
}

// ---------------------------------------------------------------------------
extern "C" void kernel_launch(void* const* d_in, const int* in_sizes, int n_in,
                              void* d_out, int out_size, void* d_ws, size_t ws_size,
                              hipStream_t stream) {
  const float* x    = (const float*)d_in[0];
  const float* Wih1 = (const float*)d_in[1];
  const float* Whh1 = (const float*)d_in[2];
  const float* bih1 = (const float*)d_in[3];
  const float* bhh1 = (const float*)d_in[4];
  const float* thr1 = (const float*)d_in[5];
  const float* Wih2 = (const float*)d_in[6];
  const float* Whh2 = (const float*)d_in[7];
  const float* bih2 = (const float*)d_in[8];
  const float* bhh2 = (const float*)d_in[9];
  const float* thr2 = (const float*)d_in[10];
  const float* Wfc  = (const float*)d_in[11];
  const float* bfc  = (const float*)d_in[12];

  // ---- workspace layout (256B aligned; all sizes already multiples of 256)
  char* ws = (char*)d_ws;
  size_t off = 0;
  auto take = [&](size_t bytes) -> void* {
    void* p = ws + off;
    off = (off + bytes + 255) & ~(size_t)255;
    return p;
  };
  _Float16* Xpad  = (_Float16*)take((size_t)Tsz * Bsz * Ipad * 2);  // 8 MB
  _Float16* Wih1p = (_Float16*)take((size_t)G4H * Ipad * 2);        // 128 KB
  _Float16* Whh1h = (_Float16*)take((size_t)G4H * Hsz * 2);
  _Float16* Wih2h = (_Float16*)take((size_t)G4H * Hsz * 2);
  _Float16* Whh2h = (_Float16*)take((size_t)G4H * Hsz * 2);
  float*    b1    = (float*)take((size_t)G4H * 4);
  float*    b2    = (float*)take((size_t)G4H * 4);
  // state region (zeroed every launch) — kept contiguous
  char* state0 = ws + off;
  float*    syn1  = (float*)take((size_t)Bsz * Hsz * 4);
  float*    mem1  = (float*)take((size_t)Bsz * Hsz * 4);
  float*    syn2  = (float*)take((size_t)Bsz * Hsz * 4);
  float*    mem2  = (float*)take((size_t)Bsz * Hsz * 4);
  float*    accum = (float*)take((size_t)Bsz * Hsz * 4);
  _Float16* m1h[2], *m2h[2];
  m1h[0] = (_Float16*)take((size_t)Bsz * Hsz * 2);
  m1h[1] = (_Float16*)take((size_t)Bsz * Hsz * 2);
  m2h[0] = (_Float16*)take((size_t)Bsz * Hsz * 2);
  m2h[1] = (_Float16*)take((size_t)Bsz * Hsz * 2);
  _Float16* spk1 = (_Float16*)take((size_t)Bsz * Hsz * 2);
  size_t state_bytes = (size_t)((ws + off) - state0);

  // ---- host-side attenuation weights: w[s] = exp(-0.05*(T-1-s)) / Z
  float w[Tsz];
  double sum = 0.0;
  for (int s = 0; s < Tsz; ++s) { w[s] = expf(-0.05f * (float)(Tsz - 1 - s)); sum += w[s]; }
  float inv = (float)(1.0 / sum);
  for (int s = 0; s < Tsz; ++s) w[s] *= inv;

  // ---- prep: zero state, convert/pad weights and x
  {
    int n = (int)(state_bytes / 4);
    zero_f32<<<(n + 255) / 256, 256, 0, stream>>>((float*)state0, n);
    prep_weights<<<4096, 256, 0, stream>>>(x, Wih1, Whh1, bih1, bhh1,
                                           Wih2, Whh2, bih2, bhh2,
                                           Xpad, Wih1p, Whh1h, Wih2h, Whh2h,
                                           b1, b2);
  }

  // ---- recurrent scan: 2 fused WMMA kernels per step
  dim3 lgrid(Hsz / 16, Bsz / 128);   // 32 x 2 blocks, 8 waves each
  for (int t = 0; t < Tsz; ++t) {
    slstm_layer1<<<lgrid, 256, 0, stream>>>(
        Xpad + (size_t)t * Bsz * Ipad, Wih1p, Whh1h, b1, thr1,
        m1h[t & 1], syn1, mem1, m1h[(t + 1) & 1], spk1);
    slstm_layer2<<<lgrid, 256, 0, stream>>>(
        spk1, Wih2h, Whh2h, b2, thr2,
        m2h[t & 1], syn2, mem2, m2h[(t + 1) & 1], accum, w[t]);
  }

  // ---- output projection
  fc_out<<<(Bsz * 8 + 255) / 256, 256, 0, stream>>>(accum, Wfc, bfc, (float*)d_out);
}
